// SuperAttention_Mutihead_16784732193001
// MI455X (gfx1250) — matrix operand order repaired and checked by reading, not compiled, not run on hardware
//
#include <hip/hip_runtime.h>

// ---------------------------------------------------------------------------
// CDNA5 (gfx1250) implementation of "SuperAttention multihead".
// All heavy math runs through v_wmma_f32_16x16x32_bf16 (wave32 WMMA).
//
// Stage 0: Kb = bf16(K)                               (one-time convert)
// Stage 1: VtT[b,h][d][j] = (Wa[h]^T @ V[b])^T        (GEMM, transA, trans-store)
// Stage 2: QW[b,h][q][j]  = Q[b] @ Wq[h]              (GEMM)
// Stage 3: flash attention: S = QW@Kb^T/16, online softmax, O += P@Vt
//          (all bf16 tile staging via GLOBAL_LOAD_ASYNC_TO_LDS_B128)
// Stage 4: out[b] = Ocat[b] @ Wo^T                    (GEMM, bf16 A, transB)
// ---------------------------------------------------------------------------

typedef __attribute__((ext_vector_type(16))) __bf16 v16bf;
typedef __attribute__((ext_vector_type(8)))  float  v8f;
typedef __attribute__((ext_vector_type(4)))  int    v4i;

union FragU {
    uint4 u[2];
    v16bf v;
};

// -------------------- async global->LDS copy (guarded) ---------------------
#if defined(__has_builtin)
#  if __has_builtin(__builtin_amdgcn_global_load_async_to_lds_b128) && \
      __has_builtin(__builtin_amdgcn_s_wait_asynccnt)
#    define HAVE_ASYNC_LDS 1
#  endif
#endif
#ifndef HAVE_ASYNC_LDS
#  define HAVE_ASYNC_LDS 0
#endif

static __device__ inline void cp16(void* lds_dst, const void* glob_src) {
#if HAVE_ASYNC_LDS
    // signature (per hipcc diagnostic): (v4i AS1*, v4i AS3*, imm offset, imm cpol)
    __builtin_amdgcn_global_load_async_to_lds_b128(
        (__attribute__((address_space(1))) v4i*)glob_src,
        (__attribute__((address_space(3))) v4i*)lds_dst,
        0, 0);
#else
    *(uint4*)lds_dst = *(const uint4*)glob_src;
#endif
}

static __device__ inline void cp_wait() {
#if HAVE_ASYNC_LDS
    __builtin_amdgcn_s_wait_asynccnt(0);
#endif
}

static __device__ inline unsigned short f2bf(float x) {
    union { float f; unsigned u; } c;
    c.f = x;
    unsigned r = c.u + 0x7FFFu + ((c.u >> 16) & 1u);   // round-to-nearest-even
    return (unsigned short)(r >> 16);
}

static __device__ inline ushort4 pack4(float4 v) {
    ushort4 h;
    h.x = f2bf(v.x); h.y = f2bf(v.y); h.z = f2bf(v.z); h.w = f2bf(v.w);
    return h;
}

static __device__ inline v8f wmma_bf16(v16bf a, v16bf b, v8f c) {
    // (neg_a, A, neg_b, B, c_mod, C, reuse_a, reuse_b)
    return __builtin_amdgcn_wmma_f32_16x16x32_bf16(false, a, false, b, (short)0, c, false, false);
}

// ---------------------------------------------------------------------------
// Stage 0: one-time fp32 -> bf16 convert (for K)
// ---------------------------------------------------------------------------
__global__ __launch_bounds__(256)
void f32_to_bf16_kernel(const float* __restrict__ in, unsigned short* __restrict__ out)
{
    long i = ((long)blockIdx.x * 256 + threadIdx.x) * 8;
    float4 a = *(const float4*)(in + i);
    float4 b = *(const float4*)(in + i + 4);
    *(ushort4*)(out + i)     = pack4(a);
    *(ushort4*)(out + i + 4) = pack4(b);
}

// ---------------------------------------------------------------------------
// Generic tiled WMMA GEMM: C[M,N] = A[M,K] * B[K,N]
//   TRA: A stored transposed in global: A[m,k] = Ag[k*lda + m]   (fp32 only)
//   TRB: B stored transposed in global: B[k,n] = Bg[n*ldb + k]   (fp32 only)
//   CT : store C transposed: C[n*ldc + m]
// Block tile 128x128, K-step 32, 256 threads = 8 waves (4 Mwaves x 2 Nwaves).
// LDS rows padded to 40 bf16 (80B) -> bank-conflict-free b128 fragment loads.
// bf16 A input is staged with async global->LDS copies.
// ---------------------------------------------------------------------------
#define BM 128
#define BN 128
#define BK 32
#define LSTR 40   // padded LDS row stride in bf16 elements (80 bytes)

template <typename TA, typename TB, typename TC, bool TRA, bool TRB, bool CT>
__global__ __launch_bounds__(256)
void gemm_wmma_kernel(const TA* __restrict__ Ag, const TB* __restrict__ Bg,
                      TC* __restrict__ Cg,
                      int M, int N, int Kd, int lda, int ldb, int ldc,
                      int H, long aSB, long aSH, long bSB, long bSH)
{
    static_assert(!TRA || sizeof(TA) == 4, "transA only for fp32");
    static_assert(!TRB || sizeof(TB) == 4, "transB only for fp32");

    __shared__ unsigned short As[BM * LSTR];
    __shared__ unsigned short Bs[BN * LSTR];

    const int z   = blockIdx.z;
    const TA* A   = Ag + (long)(z / H) * aSB + (long)(z % H) * aSH;
    const TB* B   = Bg + (long)(z / H) * bSB + (long)(z % H) * bSH;
    TC*       C   = Cg + (long)z * (long)M * (long)N;
    const int n0  = blockIdx.x * BN;
    const int m0  = blockIdx.y * BM;
    const int tid = threadIdx.x;
    const int lane = tid & 31;
    const int wid  = tid >> 5;
    const int waveM = wid >> 1;   // 0..3 -> 32 rows each
    const int waveN = wid & 1;    // 0..1 -> 64 cols each

    v8f acc[2][4] = {};

    for (int k0 = 0; k0 < Kd; k0 += BK) {
        // ------------------ stage A tile into LDS (bf16) ------------------
        if constexpr (!TRA) {
            if constexpr (sizeof(TA) == 4) {
                #pragma unroll
                for (int i = 0; i < 4; ++i) {                 // 128x32 fp32, float4 chunks
                    int chunk = tid + i * 256;
                    int r = chunk >> 3, kc = (chunk & 7) * 4;
                    float4 v = *(const float4*)((const float*)A + (long)(m0 + r) * lda + k0 + kc);
                    *(ushort4*)&As[r * LSTR + kc] = pack4(v);
                }
            } else {
                #pragma unroll
                for (int i = 0; i < 2; ++i) {                 // 128x32 bf16: async copy
                    int chunk = tid + i * 256;
                    int r = chunk >> 2, kc = (chunk & 3) * 8;
                    cp16(&As[r * LSTR + kc],
                         (const unsigned short*)A + (long)(m0 + r) * lda + k0 + kc);
                }
            }
        } else {
            #pragma unroll
            for (int i = 0; i < 4; ++i) {                     // rows are K, M contiguous
                int chunk = tid + i * 256;
                int kr = chunk >> 5, mc = (chunk & 31) * 4;
                float4 v = *(const float4*)((const float*)A + (long)(k0 + kr) * lda + m0 + mc);
                As[(mc + 0) * LSTR + kr] = f2bf(v.x);
                As[(mc + 1) * LSTR + kr] = f2bf(v.y);
                As[(mc + 2) * LSTR + kr] = f2bf(v.z);
                As[(mc + 3) * LSTR + kr] = f2bf(v.w);
            }
        }
        // -------- stage B tile into LDS, stored N-major: Bs[n][k] ---------
        if constexpr (!TRB) {
            if constexpr (sizeof(TB) == 4) {
                #pragma unroll
                for (int i = 0; i < 4; ++i) {                 // rows are K, N contiguous -> scatter
                    int chunk = tid + i * 256;
                    int kr = chunk >> 5, nc = (chunk & 31) * 4;
                    float4 v = *(const float4*)((const float*)B + (long)(k0 + kr) * ldb + n0 + nc);
                    Bs[(nc + 0) * LSTR + kr] = f2bf(v.x);
                    Bs[(nc + 1) * LSTR + kr] = f2bf(v.y);
                    Bs[(nc + 2) * LSTR + kr] = f2bf(v.z);
                    Bs[(nc + 3) * LSTR + kr] = f2bf(v.w);
                }
            } else {
                #pragma unroll
                for (int i = 0; i < 2; ++i) {
                    int chunk = tid + i * 256;
                    int kr = chunk >> 4, nc = (chunk & 15) * 8;
                    const unsigned short* bp = (const unsigned short*)B + (long)(k0 + kr) * ldb + n0 + nc;
                    #pragma unroll
                    for (int j = 0; j < 8; ++j) Bs[(nc + j) * LSTR + kr] = bp[j];
                }
            }
        } else {
            #pragma unroll
            for (int i = 0; i < 4; ++i) {                     // rows are N, K contiguous
                int chunk = tid + i * 256;
                int r = chunk >> 3, kc = (chunk & 7) * 4;
                float4 v = *(const float4*)((const float*)B + (long)(n0 + r) * ldb + k0 + kc);
                *(ushort4*)&Bs[r * LSTR + kc] = pack4(v);
            }
        }
        cp_wait();
        __syncthreads();

        // ------------------------- WMMA compute ---------------------------
        FragU a[2];
        #pragma unroll
        for (int mt = 0; mt < 2; ++mt) {
            int mr = waveM * 32 + mt * 16 + (lane & 15);
            int kc = (lane >> 4) * 8;                  // A layout: K 0-7 / 16-23 halves
            a[mt].u[0] = *(const uint4*)&As[mr * LSTR + kc];
            a[mt].u[1] = *(const uint4*)&As[mr * LSTR + kc + 16];
        }
        #pragma unroll
        for (int nt = 0; nt < 4; ++nt) {
            FragU b;
            int nr = waveN * 64 + nt * 16 + (lane & 15);
            int kc = (lane >> 4) * 8;                  // B layout: K 0-7 / 16-23 halves
            b.u[0] = *(const uint4*)&Bs[nr * LSTR + kc];
            b.u[1] = *(const uint4*)&Bs[nr * LSTR + kc + 16];
            #pragma unroll
            for (int mt = 0; mt < 2; ++mt)
                acc[mt][nt] = wmma_bf16(a[mt].v, b.v, acc[mt][nt]);
        }
        __syncthreads();
    }

    // ------------------------------ epilogue ------------------------------
    #pragma unroll
    for (int mt = 0; mt < 2; ++mt) {
        #pragma unroll
        for (int nt = 0; nt < 4; ++nt) {
            int mbase = m0 + waveM * 32 + mt * 16 + (lane >> 4) * 8;
            int n     = n0 + waveN * 64 + nt * 16 + (lane & 15);
            #pragma unroll
            for (int r = 0; r < 8; ++r) {
                int m = mbase + r;
                float val = acc[mt][nt][r];
                long idx = CT ? ((long)n * ldc + m) : ((long)m * ldc + n);
                if constexpr (sizeof(TC) == 4) ((float*)C)[idx] = val;
                else ((unsigned short*)C)[idx] = f2bf(val);
            }
        }
    }
}

// ---------------------------------------------------------------------------
// Flash attention: per (b,h), 128 q-rows per block, k-tiles of 64.
//   QW  : bf16 [bh][1024][256]
//   Kb  : bf16 [b][1024][256]
//   VtT : bf16 [bh][256][1024]    (d-major -> B fragments are K-contiguous)
//   Ocat: bf16 [b][1024][2048]    (head h occupies columns h*256..)
// 8 waves; wave w owns q-rows [w*16, w*16+16) => softmax fully intra-wave.
// All tile staging via async global->LDS B128 copies (ASYNCcnt).
// Dynamic LDS: QW 128x264 + K 64x264 + Vt 256x72 + P 8x16x72 bf16 = 156672 B.
// ---------------------------------------------------------------------------
#define QSTR 264   // 528B rows (33 banks) - conflict free
#define VSTR 72    // 144B rows

__global__ __launch_bounds__(256)
void flash_attn_kernel(const unsigned short* __restrict__ QW,
                       const unsigned short* __restrict__ Kb,
                       const unsigned short* __restrict__ VtT,
                       unsigned short* __restrict__ Ocat,
                       float scale)
{
    extern __shared__ unsigned short smem[];
    unsigned short* qs = smem;                 // 128*264
    unsigned short* ks = qs + 128 * QSTR;      // 64*264
    unsigned short* vs = ks + 64 * QSTR;       // 256*72
    unsigned short* ps = vs + 256 * VSTR;      // 8*16*72

    const int bh = blockIdx.y;
    const int b = bh >> 3, h = bh & 7;
    const int q0 = blockIdx.x * 128;
    const int tid = threadIdx.x, lane = tid & 31, wid = tid >> 5;

    const unsigned short* qwg = QW + (long)bh * 262144 + (long)q0 * 256;
    const unsigned short* kg  = Kb + (long)b * 262144;
    const unsigned short* vtg = VtT + (long)bh * 262144;

    // async-load the whole QW tile once
    #pragma unroll
    for (int i = 0; i < 16; ++i) {
        int chunk = tid + i * 256;                 // 4096 x 8 elems
        int r = chunk >> 5, kc = (chunk & 31) * 8;
        cp16(&qs[r * QSTR + kc], qwg + (long)r * 256 + kc);
    }

    float mstate[8], lsum[8];
    #pragma unroll
    for (int r = 0; r < 8; ++r) { mstate[r] = -1e30f; lsum[r] = 0.f; }
    v8f oacc[16] = {};

    const int mrow = wid * 16 + (lane & 15);       // this wave's A-rows in qs

    for (int n0 = 0; n0 < 1024; n0 += 64) {
        __syncthreads();
        // K tile: 64 x 256 bf16 (rows are sequence positions = S cols)
        #pragma unroll
        for (int i = 0; i < 8; ++i) {
            int chunk = tid + i * 256;             // 2048 x 8 elems
            int r = chunk >> 5, kc = (chunk & 31) * 8;
            cp16(&ks[r * QSTR + kc], kg + (long)(n0 + r) * 256 + kc);
        }
        // Vt tile: rows d (256) x 64 seq, already d-major bf16
        #pragma unroll
        for (int i = 0; i < 8; ++i) {
            int chunk = tid + i * 256;             // 2048 x 8 elems
            int r = chunk >> 3, jc = (chunk & 7) * 8;
            cp16(&vs[r * VSTR + jc], vtg + (long)r * 1024 + n0 + jc);
        }
        cp_wait();            // own async copies landed (incl. QW on iter 0)
        __syncthreads();      // ... collectively

        // ---- S strip = QW[16 rows] @ K^T[64 cols], K-dim 256 = 8 steps ----
        v8f sacc[4] = {};
        #pragma unroll
        for (int kk = 0; kk < 8; ++kk) {
            FragU a;
            int kc = kk * 32 + (lane >> 4) * 8;
            a.u[0] = *(const uint4*)&qs[mrow * QSTR + kc];
            a.u[1] = *(const uint4*)&qs[mrow * QSTR + kc + 16];
            #pragma unroll
            for (int nt = 0; nt < 4; ++nt) {
                FragU bfr;
                int nr = nt * 16 + (lane & 15);
                int kcb = kk * 32 + (lane >> 4) * 8;
                bfr.u[0] = *(const uint4*)&ks[nr * QSTR + kcb];
                bfr.u[1] = *(const uint4*)&ks[nr * QSTR + kcb + 16];
                sacc[nt] = wmma_bf16(a.v, bfr.v, sacc[nt]);
            }
        }
        #pragma unroll
        for (int nt = 0; nt < 4; ++nt)
            #pragma unroll
            for (int r = 0; r < 8; ++r) sacc[nt][r] *= scale;

        // -------- online softmax (rows live in 16-lane half-groups) -------
        float corr[8];
        #pragma unroll
        for (int r = 0; r < 8; ++r) {
            float tm = fmaxf(fmaxf(sacc[0][r], sacc[1][r]), fmaxf(sacc[2][r], sacc[3][r]));
            tm = fmaxf(tm, __shfl_xor(tm, 1));
            tm = fmaxf(tm, __shfl_xor(tm, 2));
            tm = fmaxf(tm, __shfl_xor(tm, 4));
            tm = fmaxf(tm, __shfl_xor(tm, 8));
            float mnew = fmaxf(mstate[r], tm);
            corr[r] = __expf(mstate[r] - mnew);
            mstate[r] = mnew;
            float rs = 0.f;
            #pragma unroll
            for (int nt = 0; nt < 4; ++nt) {
                float p = __expf(sacc[nt][r] - mnew);
                sacc[nt][r] = p;
                rs += p;
            }
            rs += __shfl_xor(rs, 1);
            rs += __shfl_xor(rs, 2);
            rs += __shfl_xor(rs, 4);
            rs += __shfl_xor(rs, 8);
            lsum[r] = lsum[r] * corr[r] + rs;
        }
        #pragma unroll
        for (int dt = 0; dt < 16; ++dt)
            #pragma unroll
            for (int r = 0; r < 8; ++r) oacc[dt][r] *= corr[r];

        // ---- P: D-layout regs -> per-wave LDS scratch -> A-layout frags ---
        unsigned short* pw = ps + wid * 16 * VSTR;
        #pragma unroll
        for (int nt = 0; nt < 4; ++nt)
            #pragma unroll
            for (int r = 0; r < 8; ++r) {
                int m = r + (lane >> 4) * 8;
                int n = nt * 16 + (lane & 15);
                pw[m * VSTR + n] = f2bf(sacc[nt][r]);
            }
        // same-wave DS ops are in-order; no barrier needed for pw reuse
        #pragma unroll
        for (int k2 = 0; k2 < 2; ++k2) {
            FragU a;
            int m = lane & 15;
            int kc = k2 * 32 + (lane >> 4) * 8;
            a.u[0] = *(const uint4*)&pw[m * VSTR + kc];
            a.u[1] = *(const uint4*)&pw[m * VSTR + kc + 16];
            #pragma unroll
            for (int dt = 0; dt < 16; ++dt) {
                FragU bfr;
                int dr = dt * 16 + (lane & 15);
                int kcb = k2 * 32 + (lane >> 4) * 8;
                bfr.u[0] = *(const uint4*)&vs[dr * VSTR + kcb];
                bfr.u[1] = *(const uint4*)&vs[dr * VSTR + kcb + 16];
                oacc[dt] = wmma_bf16(a.v, bfr.v, oacc[dt]);
            }
        }
    }

    // ------------------------- finalize & store --------------------------
    #pragma unroll
    for (int r = 0; r < 8; ++r) {
        float inv = 1.f / lsum[r];
        int q = q0 + wid * 16 + r + (lane >> 4) * 8;
        unsigned short* orow = Ocat + (long)(b * 1024 + q) * 2048 + h * 256;
        #pragma unroll
        for (int dt = 0; dt < 16; ++dt) {
            int d = dt * 16 + (lane & 15);
            orow[d] = f2bf(oacc[dt][r] * inv);
        }
    }
}

// ---------------------------------------------------------------------------
extern "C" void kernel_launch(void* const* d_in, const int* in_sizes, int n_in,
                              void* d_out, int out_size, void* d_ws, size_t ws_size,
                              hipStream_t stream)
{
    const float* Q  = (const float*)d_in[0];   // [8,1024,256]
    const float* K  = (const float*)d_in[1];   // [8,1024,256]
    const float* V  = (const float*)d_in[2];   // [8,1024,256]
    const float* Wq = (const float*)d_in[3];   // [8,256,256]
    const float* Wa = (const float*)d_in[4];   // [8,1024,1024]
    const float* Wo = (const float*)d_in[5];   // [256,2048]
    float* out = (float*)d_out;                // [8,1024,256]

    char* ws = (char*)d_ws;                    // 100 MB used
    unsigned short* VtT  = (unsigned short*)(ws);               // 64 * [256][1024] bf16
    unsigned short* QWp  = (unsigned short*)(ws + 33554432);    // 64 * [1024][256] bf16
    unsigned short* Ocat = (unsigned short*)(ws + 67108864);    // 8 * [1024][2048] bf16
    unsigned short* Kb   = (unsigned short*)(ws + 100663296);   // 8 * [1024][256] bf16

    // Stage 0: K -> bf16 (2M elements, 8 per thread)
    f32_to_bf16_kernel<<<dim3(1024), 256, 0, stream>>>(K, Kb);

    // Stage 1: VtT[bh] = (Wa[h]^T @ V[b])^T   M=1024(j) N=256(d) K=1024(i)
    gemm_wmma_kernel<float, float, unsigned short, true, false, true>
        <<<dim3(2, 8, 64), 256, 0, stream>>>(
            Wa, V, VtT, 1024, 256, 1024, /*lda*/1024, /*ldb*/256, /*ldc*/1024,
            /*H*/8, /*aSB*/0, /*aSH*/1048576, /*bSB*/262144, /*bSH*/0);

    // Stage 2: QW[bh] = Q[b] @ Wq[h]          M=1024(q) N=256(j) K=256(i)
    gemm_wmma_kernel<float, float, unsigned short, false, false, false>
        <<<dim3(2, 8, 64), 256, 0, stream>>>(
            Q, Wq, QWp, 1024, 256, 256, /*lda*/256, /*ldb*/256, /*ldc*/256,
            /*H*/8, /*aSB*/262144, /*aSH*/0, /*bSB*/0, /*bSH*/65536);

    // Stage 3: flash attention -> Ocat (bf16, heads concatenated)
    flash_attn_kernel<<<dim3(8, 64), 256, 156672, stream>>>(
        QWp, Kb, VtT, Ocat, 1.0f / 16.0f);

    // Stage 4: out[b] = Ocat[b] @ Wo^T        M=1024(q) N=256(o) K=2048(c)
    gemm_wmma_kernel<unsigned short, float, float, false, true, false>
        <<<dim3(2, 8, 8), 256, 0, stream>>>(
            Ocat, Wo, out, 1024, 256, 2048, /*lda*/2048, /*ldb*/2048, /*ldc*/256,
            /*H*/1, /*aSB*/2097152, /*aSH*/0, /*bSB*/0, /*bSH*/0);

    (void)in_sizes; (void)n_in; (void)out_size; (void)ws_size;
}
